// RNN_Model_85753317032330
// MI455X (gfx1250) — matrix operand order repaired and checked by reading, not compiled, not run on hardware
//
#include <hip/hip_runtime.h>

// ---------------------------------------------------------------------------
// RNN tree model for MI455X (gfx1250, wave32, WMMA).
//   B=32, L=4096, D=300, V=50000, C=3
// GEMM chain done with v_wmma_f32_16x16x32_bf16; K padded 300->320 (10 steps
// of 32), E padded 300->320 (20 tiles of 16). Projection (D->3) fused into
// the GEMM epilogue. Workspace: bf16 Wl (320x320) + fp32 h ping-pong buffers.
// ---------------------------------------------------------------------------

typedef __bf16 v16bf __attribute__((ext_vector_type(16)));
typedef __bf16 v8bf  __attribute__((ext_vector_type(8)));
typedef float  v8f   __attribute__((ext_vector_type(8)));

#define DD    300          // true D
#define KP    320          // padded K (10 * 32)
#define EP    320          // padded E (20 * 16)
#define LL    4096
#define BB    32
#define NOUT  8191         // 4096+2048+...+1

__device__ __forceinline__ unsigned short f2bf(float f) {
    union { float f; unsigned int u; } v; v.f = f;
    unsigned int u = v.u;
    unsigned int r = u + 0x7FFFu + ((u >> 16) & 1u);   // round-to-nearest-even
    return (unsigned short)(r >> 16);
}

// --------------------------- Wl -> bf16 padded -----------------------------
__global__ __launch_bounds__(256) void prep_wl_kernel(const float* __restrict__ Wl,
                                                      unsigned short* __restrict__ WlB) {
    int idx = blockIdx.x * 256 + threadIdx.x;          // < EP*KP = 102400
    int e = idx / KP, k = idx - e * KP;
    float v = (e < DD && k < DD) ? Wl[e * DD + k] : 0.0f;
    WlB[idx] = f2bf(v);
}

// ----------------- level 0: gather rows + projection -----------------------
__global__ __launch_bounds__(128) void gather_proj_kernel(const int*   __restrict__ ids,
                                                          const float* __restrict__ emb,
                                                          const float* __restrict__ Wp,
                                                          const float* __restrict__ bp,
                                                          float* __restrict__ h0,
                                                          float* __restrict__ out) {
    int lane = threadIdx.x & 31;
    int w    = threadIdx.x >> 5;
    int m    = blockIdx.x * 4 + w;                     // row in [0, B*L)
    int b    = m >> 12;                                // / L
    int n    = m & (LL - 1);
    long long row = ids[m];
    const float* src = emb + row * DD;
    float* dst = h0 + (size_t)m * DD;
    float p0 = 0.f, p1 = 0.f, p2 = 0.f;
    for (int k = lane; k < DD; k += 32) {
        float v = src[k];
        dst[k] = v;
        p0 += v * Wp[k];
        p1 += v * Wp[DD + k];
        p2 += v * Wp[2 * DD + k];
    }
    #pragma unroll
    for (int off = 16; off >= 1; off >>= 1) {
        p0 += __shfl_xor(p0, off, 32);
        p1 += __shfl_xor(p1, off, 32);
        p2 += __shfl_xor(p2, off, 32);
    }
    if (lane == 0) {
        float* o = out + ((size_t)b * NOUT + n) * 3;
        o[0] = p0 + bp[0];
        o[1] = p1 + bp[1];
        o[2] = p2 + bp[2];
    }
}

// ---- one tree level: h_out = relu((h[2m]+h[2m+1]) @ Wl^T + 2bl), + proj ---
// grid.x = (B*Nh)/16 M-tiles, 128 threads (4 waves). Wave w owns e-tiles
// {w, w+4, w+8, w+12, w+16}. A-fragments come from an LDS-staged bf16 x-tile.
__global__ __launch_bounds__(128) void level_kernel(const float* __restrict__ hIn,
                                                    float*       __restrict__ hOut,
                                                    const unsigned short* __restrict__ WlB,
                                                    const float* __restrict__ bl,
                                                    const float* __restrict__ Wp,
                                                    const float* __restrict__ bp,
                                                    float* __restrict__ out,
                                                    int Nh, int lvlOff) {
    __shared__ __align__(32) unsigned short ldsX[16 * KP];   // 10 KB
    __shared__ float projacc[48];                            // [16 rows][3 ch]

    const int tid   = threadIdx.x;
    const int lane  = tid & 31;
    const int w     = tid >> 5;
    const int mtile = blockIdx.x;

    // ---- stage x = h[2m] + h[2m+1] into LDS as bf16, zero-pad K ----
    for (int idx = tid; idx < 16 * KP; idx += 128) {
        int r = idx / KP;
        int k = idx - r * KP;
        size_t m = (size_t)mtile * 16 + r;
        float v = 0.0f;
        if (k < DD) {
            const float* p = hIn + 2 * m * DD + k;
            v = p[0] + p[DD];
        }
        ldsX[idx] = f2bf(v);
    }
    if (tid < 48) projacc[tid] = 0.0f;
    __syncthreads();

    // ---- WMMA main loop: 5 e-tiles per wave, 10 k-steps ----
    const int mA  = lane & 15;                     // A-frag row
    const int kbA = (lane < 16) ? 0 : 16;          // A-frag K base
    const int nB  = lane & 15;                     // B-frag column
    const int kbB = (lane < 16) ? 0 : 16;          // B-frag K base

    v8f acc[5];
    #pragma unroll
    for (int j = 0; j < 5; ++j) acc[j] = (v8f)0.0f;

    #pragma unroll
    for (int kk = 0; kk < KP / 32; ++kk) {
        const int k0 = kk * 32;
        // A fragment (contiguous K: elems 0..15 = K kbA..kbA+15)
        v8bf lo = *reinterpret_cast<const v8bf*>(&ldsX[mA * KP + k0 + kbA]);
        v8bf hi = *reinterpret_cast<const v8bf*>(&ldsX[mA * KP + k0 + kbA + 8]);
        v16bf afrag = __builtin_shufflevector(lo, hi,
                        0,1,2,3,4,5,6,7,8,9,10,11,12,13,14,15);
        #pragma unroll
        for (int j = 0; j < 5; ++j) {
            const int e0 = (w + 4 * j) * 16;
            // B fragment: 16 contiguous bf16 of Wl row (e0+nB), 32B aligned
            v16bf bfrag = *reinterpret_cast<const v16bf*>(
                WlB + (size_t)(e0 + nB) * KP + k0 + kbB);
            acc[j] = __builtin_amdgcn_wmma_f32_16x16x32_bf16(
                         false, afrag, false, bfrag, (short)0, acc[j],
                         false, false);
        }
    }

    // ---- epilogue: bias + relu, store h_out, fused projection ----
    float ps[8][3];
    #pragma unroll
    for (int r = 0; r < 8; ++r) { ps[r][0] = 0.f; ps[r][1] = 0.f; ps[r][2] = 0.f; }

    const int rbase = (lane >> 4) << 3;            // 0 for lanes 0-15, 8 for 16-31
    #pragma unroll
    for (int j = 0; j < 5; ++j) {
        const int e = (w + 4 * j) * 16 + nB;
        const bool ein = (e < DD);
        float b2 = 0.f, wp0 = 0.f, wp1 = 0.f, wp2 = 0.f;
        if (ein) {
            b2  = 2.0f * bl[e];
            wp0 = Wp[e];
            wp1 = Wp[DD + e];
            wp2 = Wp[2 * DD + e];
        }
        #pragma unroll
        for (int r = 0; r < 8; ++r) {
            float v = fmaxf(acc[j][r] + b2, 0.0f);
            if (ein) {
                size_t m = (size_t)mtile * 16 + rbase + r;
                hOut[m * DD + e] = v;
                ps[r][0] += v * wp0;
                ps[r][1] += v * wp1;
                ps[r][2] += v * wp2;
            }
        }
    }

    // reduce over the 16 columns of each half-wave (rows differ per half)
    #pragma unroll
    for (int off = 8; off >= 1; off >>= 1) {
        #pragma unroll
        for (int r = 0; r < 8; ++r) {
            ps[r][0] += __shfl_xor(ps[r][0], off, 32);
            ps[r][1] += __shfl_xor(ps[r][1], off, 32);
            ps[r][2] += __shfl_xor(ps[r][2], off, 32);
        }
    }
    if ((lane & 15) == 0) {
        #pragma unroll
        for (int r = 0; r < 8; ++r) {
            atomicAdd(&projacc[(rbase + r) * 3 + 0], ps[r][0]);
            atomicAdd(&projacc[(rbase + r) * 3 + 1], ps[r][1]);
            atomicAdd(&projacc[(rbase + r) * 3 + 2], ps[r][2]);
        }
    }
    __syncthreads();

    if (tid < 48) {
        int r = tid / 3, c = tid - r * 3;
        size_t m = (size_t)mtile * 16 + r;
        int b = (int)(m / Nh);
        int i = (int)(m - (size_t)b * Nh);
        out[((size_t)b * NOUT + lvlOff + i) * 3 + c] = projacc[tid] + bp[c];
    }
}

// ---------------------------------------------------------------------------
extern "C" void kernel_launch(void* const* d_in, const int* in_sizes, int n_in,
                              void* d_out, int out_size, void* d_ws, size_t ws_size,
                              hipStream_t stream) {
    const int*   word_ids = (const int*)  d_in[0];   // (B, L)
    const float* embedding= (const float*)d_in[1];   // (V, D)
    const float* Wl       = (const float*)d_in[2];   // (D, D)
    const float* bl       = (const float*)d_in[3];   // (D,)
    const float* Wp       = (const float*)d_in[4];   // (C, D)
    const float* bp       = (const float*)d_in[5];   // (C,)
    float* out = (float*)d_out;                      // (B, 8191, 3)

    // workspace layout (assumes ws_size >= ~237 MB):
    //   [0, 204800)        bf16 Wl padded 320x320
    //   [offA, +157286400) h ping buffer (B*L*D fp32)
    //   [offB, + 78643200) h pong buffer (B*L/2*D fp32)
    char* ws = (char*)d_ws;
    unsigned short* WlB = (unsigned short*)ws;
    const size_t offA = 262144;
    const size_t offB = offA + (size_t)BB * LL * DD * sizeof(float);
    float* hA = (float*)(ws + offA);
    float* hB = (float*)(ws + offB);

    // 1) Wl -> bf16 padded
    prep_wl_kernel<<<(EP * KP) / 256, 256, 0, stream>>>(Wl, WlB);

    // 2) gather + level-0 projection
    gather_proj_kernel<<<(BB * LL) / 4, 128, 0, stream>>>(
        word_ids, embedding, Wp, bp, hA, out);

    // 3) 12 tree levels, ping-pong h buffers
    float* hIn = hA;
    float* hOut = hB;
    int lvlOff = LL;
    for (int l = 1; l <= 12; ++l) {
        int Nh = LL >> l;                      // output rows per batch
        int Mtiles = (BB * Nh) / 16;           // M always multiple of 16
        level_kernel<<<Mtiles, 128, 0, stream>>>(
            hIn, hOut, WlB, bl, Wp, bp, out, Nh, lvlOff);
        float* t = hIn; hIn = hOut; hOut = t;
        lvlOff += Nh;
    }
}
